// MambaG2G_34514357190725
// MI455X (gfx1250) — matrix operand order repaired and checked by reading, not compiled, not run on hardware
//
#include <hip/hip_runtime.h>
#include <hip/hip_bf16.h>

// ---------------------------------------------------------------------------
// Mamba block + head for MI455X (gfx1250), wave32, WMMA bf16 path.
// Pipeline:
//   1) cvt x, W_in -> bf16
//   2) xz = x @ W_in^T                  (WMMA v_wmma_f32_16x16x32_bf16)
//   3) xc = silu(causal depthwise conv(x_in) + b)          (f32 VALU)
//   4) x_dbl = xc @ W_xproj^T  (N=48, scalar f32)
//   5) delta = softplus(dt @ W_dt^T + b_dt)  (K=16, scalar f32)
//   6) fused selective scan + gating + time-mean -> ybar (64x512)
//   7) emean = ybar @ W_out^T  (tiny)
//   8) head: x = elu(tanh(emean@W_outfc^T+b)); mu; sigma
// ---------------------------------------------------------------------------

#define BATCH   64
#define SEQLEN  256
#define DMODEL  256
#define DINNER  512
#define DSTATE  16
#define DTRANK  16
#define NTOK    (BATCH * SEQLEN)          // 16384
#define N1      (2 * DINNER)              // 1024  (xz columns)
#define XDBLC   (DTRANK + 2 * DSTATE)     // 48

typedef unsigned short u16;
typedef __bf16 v16bf __attribute__((ext_vector_type(16)));
typedef unsigned short v16u __attribute__((ext_vector_type(16)));
typedef unsigned short v8u  __attribute__((ext_vector_type(8)));
typedef float v8f __attribute__((ext_vector_type(8)));

// ---------------------------- f32 -> bf16 (RNE) ----------------------------
__device__ __forceinline__ u16 f2bf(float f) {
    unsigned int u = __float_as_uint(f);
    unsigned int r = u + 0x7fffu + ((u >> 16) & 1u);
    return (u16)(r >> 16);
}

__global__ void mamba_cvt_bf16(const float* __restrict__ in,
                               u16* __restrict__ out, int n) {
    int i = blockIdx.x * blockDim.x + threadIdx.x;
    if (i < n) out[i] = f2bf(in[i]);
}

// ----------------------- GEMM: C[M,N] = A[M,K] * W[N,K]^T -----------------
// bf16 inputs, f32 accumulate via v_wmma_f32_16x16x32_bf16.
// One wave computes a 16x64 strip (4 N-tiles), reusing the A fragment.
// K multiple of 32; M multiple of 16; N multiple of 64.
__global__ __launch_bounds__(128)
void mamba_gemm_bf16(const u16* __restrict__ A, const u16* __restrict__ W,
                     float* __restrict__ C, int M, int K, int N) {
    const int wave = blockIdx.x * (blockDim.x >> 5) + (threadIdx.x >> 5);
    const int lane = threadIdx.x & 31;
    const int ntg  = N >> 6;
    const int mt   = wave / ntg;
    const int ng   = wave - mt * ntg;
    if (mt >= (M >> 4)) return;
    const int m0 = mt << 4, n0 = ng << 6;
    const int l15  = lane & 15;
    const int lo16 = (lane < 16) ? 1 : 0;

    v8f acc0 = {}, acc1 = {}, acc2 = {}, acc3 = {};

    const u16* arow = A + (size_t)(m0 + l15) * K;
    const int aoff = lo16 ? 0 : 16;   // A: contiguous K run of 16
    const int boff = lo16 ? 0 : 16;   // B: contiguous K run of 16
    const u16* b0 = W + (size_t)(n0 +  0 + l15) * K + boff;
    const u16* b1 = W + (size_t)(n0 + 16 + l15) * K + boff;
    const u16* b2 = W + (size_t)(n0 + 32 + l15) * K + boff;
    const u16* b3 = W + (size_t)(n0 + 48 + l15) * K + boff;

    for (int k = 0; k < K; k += 32) {
        __builtin_prefetch(arow + k + 32 + aoff, 0, 1);   // global_prefetch_b8

        union { v16u v; v8u h[2]; } au;
        au.h[0] = *(const v8u*)(arow + k + aoff);
        au.h[1] = *(const v8u*)(arow + k + aoff + 8);
        v16bf af = __builtin_bit_cast(v16bf, au.v);

        v16bf f0 = __builtin_bit_cast(v16bf, *(const v16u*)(b0 + k));
        v16bf f1 = __builtin_bit_cast(v16bf, *(const v16u*)(b1 + k));
        v16bf f2 = __builtin_bit_cast(v16bf, *(const v16u*)(b2 + k));
        v16bf f3 = __builtin_bit_cast(v16bf, *(const v16u*)(b3 + k));

        acc0 = __builtin_amdgcn_wmma_f32_16x16x32_bf16(false, af, false, f0,
                                                       (short)0, acc0, false, false);
        acc1 = __builtin_amdgcn_wmma_f32_16x16x32_bf16(false, af, false, f1,
                                                       (short)0, acc1, false, false);
        acc2 = __builtin_amdgcn_wmma_f32_16x16x32_bf16(false, af, false, f2,
                                                       (short)0, acc2, false, false);
        acc3 = __builtin_amdgcn_wmma_f32_16x16x32_bf16(false, af, false, f3,
                                                       (short)0, acc3, false, false);
    }

    const int rowAdd = lo16 ? 0 : 8;   // C layout: lanes 16-31 hold rows M+8..M+15
    #pragma unroll
    for (int r = 0; r < 8; ++r) {
        float* cp = C + (size_t)(m0 + rowAdd + r) * N + n0 + l15;
        cp[0]  = acc0[r];
        cp[16] = acc1[r];
        cp[32] = acc2[r];
        cp[48] = acc3[r];
    }
}

// ------------------- causal depthwise conv (W=4) + SiLU --------------------
__global__ void mamba_conv_silu(const float* __restrict__ xz,
                                const float* __restrict__ cw,
                                const float* __restrict__ cb,
                                float* __restrict__ xc) {
    int idx = blockIdx.x * blockDim.x + threadIdx.x;
    if (idx >= NTOK * DINNER) return;
    int d = idx & (DINNER - 1);
    int m = idx >> 9;                 // token index b*SEQLEN + t
    int t = m & (SEQLEN - 1);
    float acc = cb[d];
    const float* w = cw + d * 4;
    #pragma unroll
    for (int j = 0; j < 4; ++j) {
        int tt = t + j - 3;
        if (tt >= 0) acc += w[j] * xz[(size_t)(m + j - 3) * N1 + d];
    }
    float s = acc / (1.0f + expf(-acc));   // silu
    xc[idx] = s;
}

// -------------------- x_dbl = xc @ W_xproj^T (N=48, K=512) -----------------
__global__ void mamba_xdbl(const float* __restrict__ xc,
                           const float* __restrict__ Wx,
                           float* __restrict__ xdbl) {
    int idx = blockIdx.x * blockDim.x + threadIdx.x;
    if (idx >= NTOK * XDBLC) return;
    int e = idx % XDBLC;
    int m = idx / XDBLC;
    const float* a = xc + (size_t)m * DINNER;
    const float* w = Wx + (size_t)e * DINNER;
    float acc = 0.f;
    #pragma unroll 8
    for (int k = 0; k < DINNER; ++k) acc = fmaf(a[k], w[k], acc);
    xdbl[idx] = acc;
}

// ------------- delta = softplus(dt @ W_dt^T + b_dt)  (K=16) ----------------
__global__ void mamba_delta(const float* __restrict__ xdbl,
                            const float* __restrict__ Wdt,
                            const float* __restrict__ bdt,
                            float* __restrict__ delta) {
    int idx = blockIdx.x * blockDim.x + threadIdx.x;
    if (idx >= NTOK * DINNER) return;
    int d = idx & (DINNER - 1);
    int m = idx >> 9;
    const float* r = xdbl + (size_t)m * XDBLC;      // dt = first 16 cols
    const float* w = Wdt + d * DTRANK;
    float acc = bdt[d];
    #pragma unroll
    for (int k = 0; k < DTRANK; ++k) acc = fmaf(r[k], w[k], acc);
    delta[idx] = (acc > 20.0f) ? acc : log1pf(expf(acc));   // softplus
}

// ---- fused selective scan + skip + SiLU gating + time-mean -> ybar --------
// One block per batch, one thread per channel d. h[16], A[16] in VGPRs.
__global__ __launch_bounds__(DINNER)
void mamba_scan(const float* __restrict__ delta, const float* __restrict__ xc,
                const float* __restrict__ xz, const float* __restrict__ xdbl,
                const float* __restrict__ A_log, const float* __restrict__ Dp,
                float* __restrict__ ybar) {
    __shared__ float sB[DSTATE];
    __shared__ float sC[DSTATE];
    const int b = blockIdx.x;
    const int d = threadIdx.x;

    float a[DSTATE], h[DSTATE];
    #pragma unroll
    for (int n = 0; n < DSTATE; ++n) {
        a[n] = -expf(A_log[d * DSTATE + n]);
        h[n] = 0.0f;
    }
    const float Dd = Dp[d];
    float acc = 0.0f;

    for (int t = 0; t < SEQLEN; ++t) {
        const int m = b * SEQLEN + t;
        if (d < 2 * DSTATE) {
            float v = xdbl[(size_t)m * XDBLC + DTRANK + d];
            if (d < DSTATE) sB[d] = v; else sC[d - DSTATE] = v;
        }
        __syncthreads();

        const float dl = delta[(size_t)m * DINNER + d];
        const float xv = xc[(size_t)m * DINNER + d];
        const float zv = xz[(size_t)m * N1 + DINNER + d];
        const float du = dl * xv;
        float y = 0.0f;
        #pragma unroll
        for (int n = 0; n < DSTATE; ++n) {
            float dA = expf(dl * a[n]);
            h[n] = fmaf(dA, h[n], du * sB[n]);
            y = fmaf(h[n], sC[n], y);
        }
        y = fmaf(xv, Dd, y);
        float sg = zv / (1.0f + expf(-zv));   // silu(z)
        acc = fmaf(y, sg, acc);
        __syncthreads();
    }
    ybar[b * DINNER + d] = acc * (1.0f / (float)SEQLEN);
}

// --------------- emean = ybar @ W_out^T   (64 x 512 x 256) -----------------
__global__ void mamba_emean(const float* __restrict__ ybar,
                            const float* __restrict__ Wout,
                            float* __restrict__ emean) {
    int idx = blockIdx.x * blockDim.x + threadIdx.x;
    if (idx >= BATCH * DMODEL) return;
    int e = idx & (DMODEL - 1);
    int b = idx >> 8;
    const float* y = ybar + b * DINNER;
    const float* w = Wout + (size_t)e * DINNER;
    float acc = 0.f;
    #pragma unroll 8
    for (int k = 0; k < DINNER; ++k) acc = fmaf(y[k], w[k], acc);
    emean[idx] = acc;
}

// --------------- head stage 1: x = elu(tanh(e @ W_outfc^T + b)) ------------
__global__ void mamba_head1(const float* __restrict__ emean,
                            const float* __restrict__ Wfc,
                            const float* __restrict__ bfc,
                            float* __restrict__ xout) {
    int idx = blockIdx.x * blockDim.x + threadIdx.x;
    if (idx >= BATCH * 256) return;
    int o = idx & 255;
    int b = idx >> 8;
    const float* e = emean + b * DMODEL;
    const float* w = Wfc + (size_t)o * DMODEL;
    float acc = bfc[o];
    #pragma unroll 8
    for (int k = 0; k < DMODEL; ++k) acc = fmaf(e[k], w[k], acc);
    float th = tanhf(acc);
    xout[idx] = (th > 0.0f) ? th : expm1f(th);   // elu
}

// --------------- head stage 2: mu, sigma -----------------------------------
__global__ void mamba_head2(const float* __restrict__ x,
                            const float* __restrict__ Wmu,
                            const float* __restrict__ bmu,
                            const float* __restrict__ Wsg,
                            const float* __restrict__ bsg,
                            float* __restrict__ mu,
                            float* __restrict__ sigma) {
    int idx = blockIdx.x * blockDim.x + threadIdx.x;
    if (idx >= BATCH * 64) return;
    int o = idx & 63;
    int b = idx >> 6;
    const float* xv = x + b * 256;
    const float* wm = Wmu + (size_t)o * 256;
    const float* ws = Wsg + (size_t)o * 256;
    float am = bmu[o], as = bsg[o];
    #pragma unroll 8
    for (int k = 0; k < 256; ++k) {
        am = fmaf(xv[k], wm[k], am);
        as = fmaf(xv[k], ws[k], as);
    }
    mu[idx] = am;
    float el = (as > 0.0f) ? as : expm1f(as);
    sigma[idx] = el + 1.0f + 1e-14f;
}

// ---------------------------------------------------------------------------
extern "C" void kernel_launch(void* const* d_in, const int* in_sizes, int n_in,
                              void* d_out, int out_size, void* d_ws, size_t ws_size,
                              hipStream_t stream) {
    const float* x       = (const float*)d_in[0];
    const float* W_in    = (const float*)d_in[1];
    const float* conv_w  = (const float*)d_in[2];
    const float* conv_b  = (const float*)d_in[3];
    const float* W_xproj = (const float*)d_in[4];
    const float* W_dt    = (const float*)d_in[5];
    const float* b_dt    = (const float*)d_in[6];
    const float* A_log   = (const float*)d_in[7];
    const float* D_param = (const float*)d_in[8];
    const float* W_out   = (const float*)d_in[9];
    const float* W_outfc = (const float*)d_in[10];
    const float* b_outfc = (const float*)d_in[11];
    const float* W_mu    = (const float*)d_in[12];
    const float* b_mu    = (const float*)d_in[13];
    const float* W_sigma = (const float*)d_in[14];
    const float* b_sigma = (const float*)d_in[15];

    // ---- workspace carve-out ----
    char* cur = (char*)d_ws;
    auto take = [&](size_t bytes) -> void* {
        void* p = (void*)cur;
        cur += (bytes + 255) & ~(size_t)255;
        return p;
    };
    u16*   x_bf  = (u16*)  take((size_t)NTOK * DMODEL * sizeof(u16));     //  8 MB
    u16*   Wb_in = (u16*)  take((size_t)N1 * DMODEL * sizeof(u16));       // 0.5 MB
    float* xz    = (float*)take((size_t)NTOK * N1 * sizeof(float));       // 64 MB
    float* xc    = (float*)take((size_t)NTOK * DINNER * sizeof(float));   // 32 MB
    float* xdbl  = (float*)take((size_t)NTOK * XDBLC * sizeof(float));    //  3 MB
    float* delta = (float*)take((size_t)NTOK * DINNER * sizeof(float));   // 32 MB
    float* ybar  = (float*)take((size_t)BATCH * DINNER * sizeof(float));
    float* emean = (float*)take((size_t)BATCH * DMODEL * sizeof(float));

    float* out_x     = (float*)d_out;            // (64,256)
    float* out_mu    = out_x + BATCH * 256;      // (64,64)
    float* out_sigma = out_mu + BATCH * 64;      // (64,64)

    // 1) bf16 conversions
    {
        int n = NTOK * DMODEL;
        mamba_cvt_bf16<<<(n + 255) / 256, 256, 0, stream>>>(x, x_bf, n);
        n = N1 * DMODEL;
        mamba_cvt_bf16<<<(n + 255) / 256, 256, 0, stream>>>(W_in, Wb_in, n);
    }
    // 2) xz = x @ W_in^T    (WMMA)  — (16384x256)x(256x1024)
    {
        int waves = (NTOK / 16) * (N1 / 64);    // 16384 waves
        mamba_gemm_bf16<<<waves / 4, 128, 0, stream>>>(x_bf, Wb_in, xz,
                                                       NTOK, DMODEL, N1);
    }
    // 3) conv + silu
    {
        int n = NTOK * DINNER;
        mamba_conv_silu<<<(n + 255) / 256, 256, 0, stream>>>(xz, conv_w, conv_b, xc);
    }
    // 4) x_dbl
    {
        int n = NTOK * XDBLC;
        mamba_xdbl<<<(n + 255) / 256, 256, 0, stream>>>(xc, W_xproj, xdbl);
    }
    // 5) delta
    {
        int n = NTOK * DINNER;
        mamba_delta<<<(n + 255) / 256, 256, 0, stream>>>(xdbl, W_dt, b_dt, delta);
    }
    // 6) fused scan + gate + mean  -> ybar
    mamba_scan<<<BATCH, DINNER, 0, stream>>>(delta, xc, xz, xdbl, A_log, D_param, ybar);
    // 7) emean = ybar @ W_out^T
    {
        int n = BATCH * DMODEL;
        mamba_emean<<<(n + 255) / 256, 256, 0, stream>>>(ybar, W_out, emean);
    }
    // 8) head
    {
        int n = BATCH * 256;
        mamba_head1<<<(n + 255) / 256, 256, 0, stream>>>(emean, W_outfc, b_outfc, out_x);
        n = BATCH * 64;
        mamba_head2<<<(n + 255) / 256, 256, 0, stream>>>(out_x, W_mu, b_mu,
                                                         W_sigma, b_sigma,
                                                         out_mu, out_sigma);
    }
}
